// RSNNBlock_85263690760944
// MI455X (gfx1250) — matrix operand order repaired and checked by reading, not compiled, not run on hardware
//
#include <hip/hip_runtime.h>

// ---- CDNA5 (gfx1250) wave32 WMMA types ----
typedef _Float16 half4 __attribute__((ext_vector_type(4)));
typedef _Float16 half8 __attribute__((ext_vector_type(8)));
typedef _Float16 v16h  __attribute__((ext_vector_type(16)));
typedef float    v8f   __attribute__((ext_vector_type(8)));

union F16Frag {              // one WMMA A/B operand (16 f16 = 8 VGPRs)
    v16h  v;
    half8 h[2];
};

// LIF constants from the reference
#define TSTEPS      16
#define V_THRESH    1.0f

// Tile configuration
constexpr int M_BLK   = 128;   // output rows per workgroup
constexpr int N_BLK   = 64;    // output cols per workgroup
constexpr int K_TILE  = 64;    // K chunk staged in LDS (2 WMMA k-steps of 32)
constexpr int LDS_STR = 72;    // halfs per row (64 + 8 pad) = 144 B, 16B-aligned

__launch_bounds__(256)
__global__ void snn_gemm_lif_kernel(const float* __restrict__ X,   // [B, K]
                                    const float* __restrict__ W,   // [H, K]
                                    const float* __restrict__ Bv,  // [H]
                                    float* __restrict__ Out,       // [B, H]
                                    int Brows, int Kdim, int Hdim)
{
    // Double-buffered split-f16 staging: value = hi + lo
    __shared__ _Float16 laHi[2][M_BLK][LDS_STR];
    __shared__ _Float16 laLo[2][M_BLK][LDS_STR];
    __shared__ _Float16 lbHi[2][N_BLK][LDS_STR];
    __shared__ _Float16 lbLo[2][N_BLK][LDS_STR];

    const int tid  = threadIdx.x;          // 0..255  (8 wave32)
    const int wave = tid >> 5;
    const int lane = tid & 31;
    const int r    = lane & 15;            // row-in-fragment
    const int hlf  = lane >> 4;            // lane half selects K sub-range

    // grid.x = N tiles (8), grid.y = M tiles (256): blocks sharing an X tile
    // are adjacent in launch order -> X re-reads hit L2 (192 MB holds all of X)
    const int nBase = blockIdx.x * N_BLK;
    const int mBase = blockIdx.y * M_BLK;

    const int waveM = wave & 3;            // 4 waves along M
    const int waveN = wave >> 2;           // 2 waves along N
    const int mW    = waveM * 32;          // wave tile origin inside block tile
    const int nW    = waveN * 32;

    v8f acc[2][2];
#pragma unroll
    for (int mi = 0; mi < 2; ++mi)
#pragma unroll
        for (int ni = 0; ni < 2; ++ni)
            acc[mi][ni] = (v8f){0.f, 0.f, 0.f, 0.f, 0.f, 0.f, 0.f, 0.f};

    const int nKTiles = Kdim / K_TILE;     // 512/64 = 8

    // per-thread global staging registers for one K tile
    float4 ga[8];   // A: 128x64 f32 = 2048 float4 / 256 thr = 8
    float4 gb[4];   // B:  64x64 f32 = 1024 float4 / 256 thr = 4

    auto loadGlobal = [&](int kt) {
        const int k0 = kt * K_TILE;
#pragma unroll
        for (int i = 0; i < 8; ++i) {
            const int f   = tid + i * 256;
            const int row = f >> 4;                 // 16 float4 per row
            const int k4  = (f & 15) * 4;
            ga[i] = *(const float4*)&X[(size_t)(mBase + row) * Kdim + k0 + k4];
        }
#pragma unroll
        for (int i = 0; i < 4; ++i) {
            const int f   = tid + i * 256;
            const int row = f >> 4;
            const int k4  = (f & 15) * 4;
            gb[i] = *(const float4*)&W[(size_t)(nBase + row) * Kdim + k0 + k4];
        }
    };

    auto convertStore = [&](int buf) {
#pragma unroll
        for (int i = 0; i < 8; ++i) {
            const int f   = tid + i * 256;
            const int row = f >> 4;
            const int k4  = (f & 15) * 4;
            const float4 g = ga[i];
            _Float16 hx = (_Float16)g.x, hy = (_Float16)g.y,
                     hz = (_Float16)g.z, hw = (_Float16)g.w;
            half4 hi = {hx, hy, hz, hw};
            half4 lo = {(_Float16)(g.x - (float)hx), (_Float16)(g.y - (float)hy),
                        (_Float16)(g.z - (float)hz), (_Float16)(g.w - (float)hw)};
            *(half4*)&laHi[buf][row][k4] = hi;
            *(half4*)&laLo[buf][row][k4] = lo;
        }
#pragma unroll
        for (int i = 0; i < 4; ++i) {
            const int f   = tid + i * 256;
            const int row = f >> 4;
            const int k4  = (f & 15) * 4;
            const float4 g = gb[i];
            _Float16 hx = (_Float16)g.x, hy = (_Float16)g.y,
                     hz = (_Float16)g.z, hw = (_Float16)g.w;
            half4 hi = {hx, hy, hz, hw};
            half4 lo = {(_Float16)(g.x - (float)hx), (_Float16)(g.y - (float)hy),
                        (_Float16)(g.z - (float)hz), (_Float16)(g.w - (float)hw)};
            *(half4*)&lbHi[buf][row][k4] = hi;
            *(half4*)&lbLo[buf][row][k4] = lo;
        }
    };

    auto computeTile = [&](int buf) {
#pragma unroll
        for (int ks = 0; ks < K_TILE; ks += 32) {
            // A fragments (16x32 f16): lane holds row, K = hlf*8..+7 and 16+hlf*8..+7
            F16Frag aHi[2], aLo[2];
#pragma unroll
            for (int mi = 0; mi < 2; ++mi) {
                const int row = mW + mi * 16 + r;
                aHi[mi].h[0] = *(const half8*)&laHi[buf][row][ks + hlf * 8];
                aHi[mi].h[1] = *(const half8*)&laHi[buf][row][ks + 16 + hlf * 8];
                aLo[mi].h[0] = *(const half8*)&laLo[buf][row][ks + hlf * 8];
                aLo[mi].h[1] = *(const half8*)&laLo[buf][row][ks + 16 + hlf * 8];
            }
            // B fragments (32x16 f16): lane holds column (= W row), K = hlf*8..+7 and 16+hlf*8..+7
            F16Frag bHi[2], bLo[2];
#pragma unroll
            for (int ni = 0; ni < 2; ++ni) {
                const int col = nW + ni * 16 + r;
                bHi[ni].h[0] = *(const half8*)&lbHi[buf][col][ks + hlf * 8];
                bHi[ni].h[1] = *(const half8*)&lbHi[buf][col][ks + 16 + hlf * 8];
                bLo[ni].h[0] = *(const half8*)&lbLo[buf][col][ks + hlf * 8];
                bLo[ni].h[1] = *(const half8*)&lbLo[buf][col][ks + 16 + hlf * 8];
            }
            // Split-precision product: hi*hi + hi*lo + lo*hi, f32 accumulate
#pragma unroll
            for (int mi = 0; mi < 2; ++mi)
#pragma unroll
                for (int ni = 0; ni < 2; ++ni) {
                    acc[mi][ni] = __builtin_amdgcn_wmma_f32_16x16x32_f16(
                        false, aHi[mi].v, false, bHi[ni].v, (short)0, acc[mi][ni], false, false);
                    acc[mi][ni] = __builtin_amdgcn_wmma_f32_16x16x32_f16(
                        false, aHi[mi].v, false, bLo[ni].v, (short)0, acc[mi][ni], false, false);
                    acc[mi][ni] = __builtin_amdgcn_wmma_f32_16x16x32_f16(
                        false, aLo[mi].v, false, bHi[ni].v, (short)0, acc[mi][ni], false, false);
                }
        }
    };

    // ---- software pipeline: prologue fills buffer 0 ----
    loadGlobal(0);
    convertStore(0);
    __syncthreads();

    for (int kt = 0; kt < nKTiles; ++kt) {
        const int cur = kt & 1;
        const bool more = (kt + 1 < nKTiles);

        if (more) loadGlobal(kt + 1);          // global_load_b128 overlap with WMMA
        if (kt + 2 < nKTiles) {                // cache prefetch two tiles ahead
            const int row = tid >> 4, k4 = (tid & 15) * 4;
            __builtin_prefetch(&X[(size_t)(mBase + row) * Kdim + (kt + 2) * K_TILE + k4], 0, 1);
        }

        computeTile(cur);                      // WMMAs on current buffer

        if (more) convertStore(cur ^ 1);       // fill next buffer
        __syncthreads();                       // single barrier per iteration
    }

    // ---- fused LIF epilogue on the f32 accumulators ----
    // C/D layout: VGPR j -> M = hlf*8 + j, N = r (within each 16x16 tile)
#pragma unroll
    for (int ni = 0; ni < 2; ++ni) {
        const int col  = nBase + nW + ni * 16 + r;
        const float bc = Bv[col];
#pragma unroll
        for (int mi = 0; mi < 2; ++mi) {
            const int rowTile = mBase + mW + mi * 16 + hlf * 8;
#pragma unroll
            for (int j = 0; j < 8; ++j) {
                const float cur = acc[mi][ni][j] + bc;
                float v = 0.f, sacc = 0.f;
#pragma unroll
                for (int t = 0; t < TSTEPS; ++t) {
                    v = 0.5f * (v + cur);                 // v += (cur - v)/tau, tau=2
                    const float sp = (v >= V_THRESH) ? 1.f : 0.f;
                    sacc += sp;
                    v -= sp * v;                          // hard reset to 0
                }
                Out[(size_t)(rowTile + j) * Hdim + col] = sacc * (1.f / TSTEPS);
            }
        }
    }
}

extern "C" void kernel_launch(void* const* d_in, const int* in_sizes, int n_in,
                              void* d_out, int out_size, void* d_ws, size_t ws_size,
                              hipStream_t stream) {
    (void)n_in; (void)d_ws; (void)ws_size; (void)out_size;
    const float* X  = (const float*)d_in[0];   // [B, K]
    const float* W  = (const float*)d_in[1];   // [H, K]
    const float* Bv = (const float*)d_in[2];   // [H]
    float* Out = (float*)d_out;                // [B, H]

    const int Hdim  = in_sizes[2];             // 512
    const int Kdim  = in_sizes[1] / Hdim;      // 512
    const int Brows = in_sizes[0] / Kdim;      // 32768

    dim3 block(256, 1, 1);
    dim3 grid(Hdim / N_BLK, Brows / M_BLK, 1); // (8, 256): N fastest for L2 reuse of X
    snn_gemm_lif_kernel<<<grid, block, 0, stream>>>(X, W, Bv, Out, Brows, Kdim, Hdim);
}
